// _PointnetSAModuleBase_24481313587578
// MI455X (gfx1250) — matrix operand order repaired and checked by reading, not compiled, not run on hardware
//
#include <hip/hip_runtime.h>
#include <hip/hip_bf16.h>

// Problem constants (from the reference)
#define BATCH 4
#define NPTS 16384
#define NCH 64
#define NPOINT 2048
#define NSAMPLE 32
#define RAD2 0.04f
#define KIN 67   // 3 + 64 input channels
#define KPAD 96  // padded to 3 * 32 for bf16 WMMA K-steps

typedef __attribute__((ext_vector_type(16))) __bf16 v16bf;
typedef __attribute__((ext_vector_type(8)))  __bf16 v8bf;
typedef __attribute__((ext_vector_type(8)))  float  v8f;

// Weight-pack geometry: per layer, per (mtile, kt) fragment, 32 lanes x 16 bf16.
// L1: 4 mtiles x 3 kt = 12 frags ; L2: 4 x 2 = 8 ; L3: 8 x 2 = 16. 36 frags total.
#define FRAGS_L1 12
#define FRAGS_L2 8
#define FRAGS_L3 16
#define WPACK_ELEMS ((FRAGS_L1 + FRAGS_L2 + FRAGS_L3) * 32 * 16)  // 18432 bf16

// ---------------------------------------------------------------------------
// Kernel 0: pack f32 weights into bf16 WMMA A-fragment order (zero K-padding).
// A layout (16x32 bf16): lane l -> m = l%16; element e:
//   k = 16*(l/16) + e
// ---------------------------------------------------------------------------
__device__ __forceinline__ void pack_one(const float* W, int K, int KT,
                                         __bf16* dst, int f) {
  int frag  = f >> 9;          // /512
  int within = f & 511;
  int lane  = within >> 4;
  int e     = within & 15;
  int mtile = frag / KT;
  int kt    = frag - mtile * KT;
  int half  = lane >> 4;
  int m     = mtile * 16 + (lane & 15);
  int vv    = e >> 1;
  int pos   = e & 1;
  int ka    = half * 16 + e;
  int kg    = kt * 32 + ka;
  float val = (kg < K) ? W[m * K + kg] : 0.0f;
  dst[f] = (__bf16)val;
}

__global__ __launch_bounds__(256) void wpack_kernel(const float* __restrict__ W1,
                                                    const float* __restrict__ W2,
                                                    const float* __restrict__ W3,
                                                    __bf16* __restrict__ wpack) {
  int t = blockIdx.x * 256 + threadIdx.x;
  const int n1 = FRAGS_L1 * 512;           // 6144
  const int n2 = FRAGS_L2 * 512;           // 4096
  const int n3 = FRAGS_L3 * 512;           // 8192
  if (t < n1) {
    pack_one(W1, KIN, 3, wpack, t);
  } else if (t < n1 + n2) {
    pack_one(W2, 64, 2, wpack + n1, t - n1);
  } else if (t < n1 + n2 + n3) {
    pack_one(W3, 64, 2, wpack + n1 + n2, t - (n1 + n2));
  }
}

// ---------------------------------------------------------------------------
// Kernel 1: Farthest Point Sampling. One block per batch, 1024 threads,
// each thread owns 16 points in registers. 2047 serial argmax iterations.
// ---------------------------------------------------------------------------
__global__ __launch_bounds__(1024) void fps_kernel(const float* __restrict__ xyz,
                                                   int*   __restrict__ fps_idx_ws,
                                                   float* __restrict__ new_xyz_ws,
                                                   float* __restrict__ out_new_xyz,
                                                   float* __restrict__ out_fps_idx) {
  const int b   = blockIdx.x;
  const int tid = threadIdx.x;
  const float* X = xyz + (size_t)b * NPTS * 3;

  float px[16], py[16], pz[16], dist[16];
#pragma unroll
  for (int i = 0; i < 16; ++i) {
    int p = tid + i * 1024;
    px[i] = X[p * 3 + 0];
    py[i] = X[p * 3 + 1];
    pz[i] = X[p * 3 + 2];
    dist[i] = 1e10f;
  }

  __shared__ float red_v[32];
  __shared__ int   red_i[32];
  __shared__ float cur[3];

  if (tid == 0) {
    cur[0] = X[0]; cur[1] = X[1]; cur[2] = X[2];
    fps_idx_ws[b * NPOINT] = 0;
    out_fps_idx[b * NPOINT] = 0.0f;
    size_t o = (size_t)(b * NPOINT) * 3;
    new_xyz_ws[o + 0] = X[0]; new_xyz_ws[o + 1] = X[1]; new_xyz_ws[o + 2] = X[2];
    out_new_xyz[o + 0] = X[0]; out_new_xyz[o + 1] = X[1]; out_new_xyz[o + 2] = X[2];
  }
  __syncthreads();

  const int lane = tid & 31;
  const int wave = tid >> 5;

  for (int t = 1; t < NPOINT; ++t) {
    float cx = cur[0], cy = cur[1], cz = cur[2];
    float bv = -1.0f;
    int   bi = 0;
#pragma unroll
    for (int i = 0; i < 16; ++i) {
      float dx = px[i] - cx, dy = py[i] - cy, dz = pz[i] - cz;
      float d = dx * dx + dy * dy + dz * dz;
      dist[i] = fminf(dist[i], d);
      if (dist[i] > bv) { bv = dist[i]; bi = tid + i * 1024; }
    }
    // wave32 argmax reduction (lowest index wins ties)
#pragma unroll
    for (int m = 16; m >= 1; m >>= 1) {
      float ov = __shfl_xor(bv, m, 32);
      int   oi = __shfl_xor(bi, m, 32);
      if (ov > bv || (ov == bv && oi < bi)) { bv = ov; bi = oi; }
    }
    if (lane == 0) { red_v[wave] = bv; red_i[wave] = bi; }
    __syncthreads();
    if (tid < 32) {
      bv = red_v[tid]; bi = red_i[tid];
#pragma unroll
      for (int m = 16; m >= 1; m >>= 1) {
        float ov = __shfl_xor(bv, m, 32);
        int   oi = __shfl_xor(bi, m, 32);
        if (ov > bv || (ov == bv && oi < bi)) { bv = ov; bi = oi; }
      }
      if (tid == 0) {
        float nx = X[bi * 3 + 0], ny = X[bi * 3 + 1], nz = X[bi * 3 + 2];
        cur[0] = nx; cur[1] = ny; cur[2] = nz;
        fps_idx_ws[b * NPOINT + t] = bi;
        out_fps_idx[b * NPOINT + t] = (float)bi;
        size_t o = (size_t)(b * NPOINT + t) * 3;
        new_xyz_ws[o + 0] = nx; new_xyz_ws[o + 1] = ny; new_xyz_ws[o + 2] = nz;
        out_new_xyz[o + 0] = nx; out_new_xyz[o + 1] = ny; out_new_xyz[o + 2] = nz;
      }
    }
    __syncthreads();
  }
}

// ---------------------------------------------------------------------------
// Kernel 2: Ball query. One wave32 per center, 8 waves per block.
// ---------------------------------------------------------------------------
__global__ __launch_bounds__(256) void ballq_kernel(const float* __restrict__ xyz,
                                                    const float* __restrict__ new_xyz_ws,
                                                    int*   __restrict__ ball_ws,
                                                    float* __restrict__ out_ball) {
  const int wave = threadIdx.x >> 5;
  const int lane = threadIdx.x & 31;
  const int c = blockIdx.x * 8 + wave;       // center id in [0, B*NPOINT)
  const int b = c >> 11;                     // NPOINT == 2048
  const float* X = xyz + (size_t)b * NPTS * 3;

  const float cx = new_xyz_ws[c * 3 + 0];
  const float cy = new_xyz_ws[c * 3 + 1];
  const float cz = new_xyz_ws[c * 3 + 2];

  __shared__ int selbuf[8][NSAMPLE];

  int cnt = 0;
  for (int j = 0; j < NPTS && cnt < NSAMPLE; j += 32) {
    int p = j + lane;
    float dx = X[p * 3 + 0] - cx;
    float dy = X[p * 3 + 1] - cy;
    float dz = X[p * 3 + 2] - cz;
    float d2 = dx * dx + dy * dy + dz * dz;
    bool inside = d2 < RAD2;
    unsigned mask = __builtin_amdgcn_ballot_w32(inside);
    int rank = __builtin_popcount(mask & ((1u << lane) - 1u));
    int slot = cnt + rank;
    if (inside && slot < NSAMPLE) selbuf[wave][slot] = p;
    cnt += __builtin_popcount(mask);
    if (cnt > NSAMPLE) cnt = NSAMPLE;
  }
  __syncthreads();

  int v;
  if (cnt == 0) v = 0;
  else          v = selbuf[wave][lane < cnt ? lane : 0];
  ball_ws[c * NSAMPLE + lane] = v;
  out_ball[c * NSAMPLE + lane] = (float)v;
}

// ---------------------------------------------------------------------------
// Fragment-ordered LDS addressing for B matrices (K x 32, bf16).
// Fragment (kt, ntile): 32 lanes x 16 bf16, lane = nn + 16*(kk>>4), e = kk&15,
// where kk = k - 32*kt, nn = n - 16*ntile.
// ---------------------------------------------------------------------------
__device__ __forceinline__ int bfrag_addr(int k, int n) {
  int kt = k >> 5, kk = k & 31;
  int nt = n >> 4, nn = n & 15;
  int lane = nn + ((kk >> 4) << 4);
  int e = kk & 15;
  return (((kt * 2 + nt) * 32 + lane) << 4) + e;
}

// ---------------------------------------------------------------------------
// Kernel 3: gather + MLP (67->64->64->128, bias+ReLU) + max over 32 samples.
// One block (8 wave32) per center. A-fragments: one v16bf global load from the
// packed weights. B-fragments: one v16bf LDS load. Epilogue: one 16B LDS store.
// ---------------------------------------------------------------------------
__global__ __launch_bounds__(256) void mlp_kernel(const float* __restrict__ xyz,
                                                  const float* __restrict__ feats,
                                                  const float* __restrict__ B1,
                                                  const float* __restrict__ B2,
                                                  const float* __restrict__ B3,
                                                  const v16bf* __restrict__ wpackV,
                                                  const float* __restrict__ new_xyz_ws,
                                                  const int*   __restrict__ ball_ws,
                                                  float* __restrict__ out_feat) {
  const int cIdx = blockIdx.x;                 // 0 .. B*NPOINT-1
  const int b = cIdx >> 11;
  const int s = cIdx & 2047;
  const int tid = threadIdx.x;

  // X: 6 fragments (3 kt x 2 ntile); H: 4 fragments (2 kt x 2 ntile)
  __shared__ v16bf XsV[6 * 32];
  __shared__ v16bf HsV[4 * 32];
  __shared__ int   pidx[NSAMPLE];
  __shared__ float cen[3];
  __bf16* Xs = (__bf16*)XsV;
  __bf16* Hs = (__bf16*)HsV;

  if (tid < NSAMPLE) pidx[tid] = ball_ws[cIdx * NSAMPLE + tid];
  if (tid < 3)       cen[tid] = new_xyz_ws[cIdx * 3 + tid];
  __syncthreads();

  const float* X = xyz   + (size_t)b * NPTS * 3;
  const float* F = feats + (size_t)b * NCH * NPTS;

  // Gather [3 relative xyz ; 64 features ; zero pad] into fragment order
  for (int e = tid; e < KPAD * NSAMPLE; e += 256) {
    int k = e >> 5, n = e & 31;
    int p = pidx[n];
    float val;
    if (k < 3)        val = X[p * 3 + k] - cen[k];
    else if (k < KIN) val = F[(size_t)(k - 3) * NPTS + p];
    else              val = 0.0f;
    Xs[bfrag_addr(k, n)] = (__bf16)val;
  }
  __syncthreads();

  const int lane = tid & 31;
  const int wave = tid >> 5;
  const int half = lane >> 4;
  const int ln   = lane & 15;

  const v16bf* wp1 = wpackV;                  // 12 frags * 32 lanes
  const v16bf* wp2 = wpackV + FRAGS_L1 * 32;  //  8 frags
  const v16bf* wp3 = wpackV + (FRAGS_L1 + FRAGS_L2) * 32;  // 16 frags

  // ---- Layer 1: [64 x 96] x [96 x 32] -> Hs ----
  {
    const int mtile = wave >> 1, ntile = wave & 1;
    v8f c = {0.f, 0.f, 0.f, 0.f, 0.f, 0.f, 0.f, 0.f};
#pragma unroll
    for (int kt = 0; kt < 3; ++kt) {
      v16bf a  = wp1[(mtile * 3 + kt) * 32 + lane];
      v16bf bb = XsV[(kt * 2 + ntile) * 32 + lane];
      c = __builtin_amdgcn_wmma_f32_16x16x32_bf16(false, a, false, bb, (short)0, c, false, false);
    }
    // rows r=0..7 land at 16 contiguous bytes in fragment order
    v8bf h;
#pragma unroll
    for (int r = 0; r < 8; ++r) {
      float v = c[r] + B1[mtile * 16 + 8 * half + r];
      h[r] = (__bf16)(v > 0.f ? v : 0.f);
    }
    int dst = (((mtile >> 1) * 2 + ntile) * 32 + (ln + ((mtile & 1) << 4))) * 16 + 8 * half;
    *(v8bf*)(Hs + dst) = h;
  }
  __syncthreads();

  // ---- Layer 2: [64 x 64] x [64 x 32], input Hs, output into Xs (reused) ----
  {
    const int mtile = wave >> 1, ntile = wave & 1;
    v8f c = {0.f, 0.f, 0.f, 0.f, 0.f, 0.f, 0.f, 0.f};
#pragma unroll
    for (int kt = 0; kt < 2; ++kt) {
      v16bf a  = wp2[(mtile * 2 + kt) * 32 + lane];
      v16bf bb = HsV[(kt * 2 + ntile) * 32 + lane];
      c = __builtin_amdgcn_wmma_f32_16x16x32_bf16(false, a, false, bb, (short)0, c, false, false);
    }
    v8bf h;
#pragma unroll
    for (int r = 0; r < 8; ++r) {
      float v = c[r] + B2[mtile * 16 + 8 * half + r];
      h[r] = (__bf16)(v > 0.f ? v : 0.f);
    }
    int dst = (((mtile >> 1) * 2 + ntile) * 32 + (ln + ((mtile & 1) << 4))) * 16 + 8 * half;
    *(v8bf*)(Xs + dst) = h;                   // H2 fragments live in Xs
  }
  __syncthreads();

  // ---- Layer 3: [128 x 64] x [64 x 32]; each wave: one M-tile, both N-tiles;
  //      fused bias+ReLU and max over the 32 samples. ----
  {
    const int mtile = wave;
    v8f c0 = {0.f, 0.f, 0.f, 0.f, 0.f, 0.f, 0.f, 0.f};
    v8f c1 = {0.f, 0.f, 0.f, 0.f, 0.f, 0.f, 0.f, 0.f};
#pragma unroll
    for (int kt = 0; kt < 2; ++kt) {
      v16bf a  = wp3[(mtile * 2 + kt) * 32 + lane];
      v16bf b0 = XsV[(kt * 2 + 0) * 32 + lane];
      v16bf b1 = XsV[(kt * 2 + 1) * 32 + lane];
      c0 = __builtin_amdgcn_wmma_f32_16x16x32_bf16(false, a, false, b0, (short)0, c0, false, false);
      c1 = __builtin_amdgcn_wmma_f32_16x16x32_bf16(false, a, false, b1, (short)0, c1, false, false);
    }
#pragma unroll
    for (int r = 0; r < 8; ++r) {
      int m = mtile * 16 + r + 8 * half;
      float bias = B3[m];
      float v0 = c0[r] + bias; v0 = v0 > 0.f ? v0 : 0.f;
      float v1 = c1[r] + bias; v1 = v1 > 0.f ? v1 : 0.f;
      float t = fmaxf(v0, v1);
#pragma unroll
      for (int msk = 8; msk >= 1; msk >>= 1)
        t = fmaxf(t, __shfl_xor(t, msk, 32));
      if (ln == 0)
        out_feat[((size_t)b * 128 + m) * NPOINT + s] = t;
    }
  }
}

// ---------------------------------------------------------------------------
// d_out layout (floats, reference return order):
//   [0)          new_xyz        B*NPOINT*3      = 24576
//   [24576)      new_xyz_idx    B*NPOINT        = 8192
//   [32768)      new_features   B*128*NPOINT    = 1048576
//   [1081344)    idx            B*NPOINT*32     = 262144
// ws layout:
//   [0)          fps_idx  (int)    32768 B
//   [32768)      new_xyz  (float)  98304 B
//   [131072)     ball_idx (int)    1048576 B
//   [1179648)    wpack    (bf16)   36864 B   (32B-aligned)
// ---------------------------------------------------------------------------
extern "C" void kernel_launch(void* const* d_in, const int* in_sizes, int n_in,
                              void* d_out, int out_size, void* d_ws, size_t ws_size,
                              hipStream_t stream) {
  const float* xyz   = (const float*)d_in[0];
  const float* feats = (const float*)d_in[1];
  const float* W1 = (const float*)d_in[2];
  const float* B1 = (const float*)d_in[3];
  const float* W2 = (const float*)d_in[4];
  const float* B2 = (const float*)d_in[5];
  const float* W3 = (const float*)d_in[6];
  const float* B3 = (const float*)d_in[7];

  float* out = (float*)d_out;
  float* out_new_xyz = out;
  float* out_fps_idx = out + 24576;
  float* out_feat    = out + 32768;
  float* out_ball    = out + 1081344;

  char* ws = (char*)d_ws;
  int*    fps_idx_ws = (int*)ws;                   // 32768 B
  float*  new_xyz_ws = (float*)(ws + 32768);       // 98304 B
  int*    ball_ws    = (int*)(ws + 131072);        // 1048576 B
  __bf16* wpack      = (__bf16*)(ws + 1179648);    // 36864 B

  wpack_kernel<<<(WPACK_ELEMS + 255) / 256, 256, 0, stream>>>(W1, W2, W3, wpack);
  fps_kernel<<<BATCH, 1024, 0, stream>>>(xyz, fps_idx_ws, new_xyz_ws,
                                         out_new_xyz, out_fps_idx);
  ballq_kernel<<<(BATCH * NPOINT) / 8, 256, 0, stream>>>(xyz, new_xyz_ws,
                                                         ball_ws, out_ball);
  mlp_kernel<<<BATCH * NPOINT, 256, 0, stream>>>(xyz, feats, B1, B2, B3,
                                                 (const v16bf*)wpack,
                                                 new_xyz_ws, ball_ws, out_feat);
}
